// MEGNet_EdgeConvolution_7499012898893
// MI455X (gfx1250) — matrix operand order repaired and checked by reading, not compiled, not run on hardware
//
#include <hip/hip_runtime.h>

typedef __attribute__((ext_vector_type(16))) _Float16 v16h;
typedef __attribute__((ext_vector_type(8)))  _Float16 v8h;
typedef __attribute__((ext_vector_type(8)))  float    v8f;

#define BN_EPS 1e-5f

#define WMMA_F16(A,B,C) __builtin_amdgcn_wmma_f32_16x16x32_f16(false,(A),false,(B),(short)0,(C),false,false)

static __device__ __forceinline__ v16h pack16(v8h a, v8h b) {
  v16h r;
#pragma unroll
  for (int i = 0; i < 8; ++i) { r[i] = a[i]; r[i + 8] = b[i]; }
  return r;
}

// A fragment (16x32 f16, MxK) from row-major f16 rows.
// Per ISA layout: lane L holds row M=L%16; halves 0..7 -> K=kb+off+0..7,
// halves 8..15 -> K=kb+off+16..23, off = (L<16 ? 0 : 8).
static __device__ __forceinline__ v16h load_a_f16(const _Float16* __restrict__ p) {
  v8h a = *(const v8h*)p;
  v8h b = *(const v8h*)(p + 16);
  return pack16(a, b);
}

// Same A mapping but from f32 memory, converting to f16 on the fly.
static __device__ __forceinline__ v16h load_a_f32(const float* __restrict__ p) {
  float4 x0 = ((const float4*)p)[0];
  float4 x1 = ((const float4*)p)[1];
  float4 y0 = ((const float4*)(p + 16))[0];
  float4 y1 = ((const float4*)(p + 16))[1];
  v16h r;
  r[0]=(_Float16)x0.x; r[1]=(_Float16)x0.y; r[2]=(_Float16)x0.z; r[3]=(_Float16)x0.w;
  r[4]=(_Float16)x1.x; r[5]=(_Float16)x1.y; r[6]=(_Float16)x1.z; r[7]=(_Float16)x1.w;
  r[8]=(_Float16)y0.x; r[9]=(_Float16)y0.y; r[10]=(_Float16)y0.z; r[11]=(_Float16)y0.w;
  r[12]=(_Float16)y1.x; r[13]=(_Float16)y1.y; r[14]=(_Float16)y1.z; r[15]=(_Float16)y1.w;
  return r;
}

// B fragments are pre-swizzled in memory: frag f -> 32 lanes x 16 contiguous halves.
static __device__ __forceinline__ v16h load_b_frag(const _Float16* __restrict__ wf, int f, int lane) {
  const _Float16* p = wf + ((size_t)f << 9) + (lane << 4);
  v8h a = *(const v8h*)p;
  v8h b = *(const v8h*)(p + 8);
  return pack16(a, b);
}

// Convert W1 [256x64] f32 (row-major, K x N) into B-fragment-ordered f16.
// B layout (32x16 KxN per frag): lane L: N = nt*16 + L%16; half h -> K = ks*32 + (L<16?0:8) + h + (h&8).
__global__ __launch_bounds__(256) void megnet_prep_w1(const float* __restrict__ W1,
                                                      _Float16* __restrict__ wf) {
  for (int i = threadIdx.x; i < 16384; i += 256) {
    int h = i & 15, l = (i >> 4) & 31, f = i >> 9;
    int ks = f >> 2, nt = f & 3;
    int k = ks * 32 + ((l < 16) ? 0 : 8) + h + (h & 8);
    int n = nt * 16 + (l & 15);
    wf[i] = (_Float16)W1[k * 64 + n];
  }
}

// Layer 1: gather-concat [src|dest|edge|global[batch]] (E x 256 f32) @ W1 -> relu -> act1 (f16)
// plus per-block partial BN stats (deterministic, no atomics).
__global__ __launch_bounds__(256) void megnet_gemm1(
    const float* __restrict__ src, const float* __restrict__ dst,
    const float* __restrict__ edg, const float* __restrict__ gattr,
    const int* __restrict__ batch,
    const _Float16* __restrict__ wf, const float* __restrict__ bias,
    _Float16* __restrict__ actOut, float* __restrict__ partials, int E) {
  __shared__ float part[8][4][32][2];
  const int tid = threadIdx.x, wv = tid >> 5, lane = tid & 31;
  const int lrow = lane & 15, off8 = (lane < 16) ? 0 : 8;
  for (int i = tid; i < 2048; i += 256) ((float*)part)[i] = 0.f;
  __syncthreads();

  const long long job = ((long long)blockIdx.x * 8 + wv) * 64LL;
  if (job < E) {
    v8f acc[4][4];
    v8f z = {};
#pragma unroll
    for (int rt = 0; rt < 4; ++rt)
#pragma unroll
      for (int nt = 0; nt < 4; ++nt) acc[rt][nt] = z;

#pragma unroll
    for (int ks = 0; ks < 8; ++ks) {
      const int sel = ks >> 1;
      const int cb = (ks & 1) * 32;
      v16h bf[4];
#pragma unroll
      for (int nt = 0; nt < 4; ++nt) bf[nt] = load_b_frag(wf, ks * 4 + nt, lane);
#pragma unroll
      for (int rt = 0; rt < 4; ++rt) {
        long long row = job + rt * 16 + lrow;
        if (row >= E) row = E - 1;
        const float* base;
        if (sel == 0)      base = src + row * 64;
        else if (sel == 1) base = dst + row * 64;
        else if (sel == 2) base = edg + row * 64;
        else               base = gattr + (long long)batch[row] * 64;
        v16h a = load_a_f32(base + cb + off8);
#pragma unroll
        for (int nt = 0; nt < 4; ++nt) acc[rt][nt] = WMMA_F16(a, bf[nt], acc[rt][nt]);
      }
    }

    float bn[4];
#pragma unroll
    for (int nt = 0; nt < 4; ++nt) bn[nt] = bias[nt * 16 + lrow];
#pragma unroll
    for (int nt = 0; nt < 4; ++nt) {
      float ls = 0.f, lq = 0.f;
#pragma unroll
      for (int rt = 0; rt < 4; ++rt) {
#pragma unroll
        for (int r = 0; r < 8; ++r) {
          long long row = job + rt * 16 + r + off8;  // D: lanes 16-31 hold M = r+8
          float v = fmaxf(acc[rt][nt][r] + bn[nt], 0.f);
          if (row < E) {
            actOut[row * 64 + nt * 16 + lrow] = (_Float16)v;
            ls += v; lq += v * v;
          }
        }
      }
      part[wv][nt][lane][0] = ls;
      part[wv][nt][lane][1] = lq;
    }
  }
  __syncthreads();
  if (tid < 128) {
    const int s = tid >> 6, n = tid & 63, nt = n >> 4, lr = n & 15;
    float a = 0.f;
#pragma unroll
    for (int w = 0; w < 8; ++w) a += part[w][nt][lr][s] + part[w][nt][lr + 16][s];
    partials[(size_t)blockIdx.x * 128 + tid] = a;
  }
}

// Layers 2/3: act(f16, Ex64) @ folded-W -> relu -> actOut (f16 or f32) + partial stats.
template <bool OUT_F32>
__global__ __launch_bounds__(256) void megnet_gemm_mid(
    const _Float16* __restrict__ actIn, const _Float16* __restrict__ wf,
    const float* __restrict__ bias, void* __restrict__ actOutV,
    float* __restrict__ partials, int E) {
  __shared__ float part[8][4][32][2];
  const int tid = threadIdx.x, wv = tid >> 5, lane = tid & 31;
  const int lrow = lane & 15, off8 = (lane < 16) ? 0 : 8;
  for (int i = tid; i < 2048; i += 256) ((float*)part)[i] = 0.f;
  __syncthreads();

  const long long job = ((long long)blockIdx.x * 8 + wv) * 64LL;
  if (job < E) {
    v8f acc[4][4];
    v8f z = {};
#pragma unroll
    for (int rt = 0; rt < 4; ++rt)
#pragma unroll
      for (int nt = 0; nt < 4; ++nt) acc[rt][nt] = z;

#pragma unroll
    for (int ks = 0; ks < 2; ++ks) {
      v16h bf[4];
#pragma unroll
      for (int nt = 0; nt < 4; ++nt) bf[nt] = load_b_frag(wf, ks * 4 + nt, lane);
#pragma unroll
      for (int rt = 0; rt < 4; ++rt) {
        long long row = job + rt * 16 + lrow;
        if (row >= E) row = E - 1;
        v16h a = load_a_f16(actIn + row * 64 + ks * 32 + off8);
#pragma unroll
        for (int nt = 0; nt < 4; ++nt) acc[rt][nt] = WMMA_F16(a, bf[nt], acc[rt][nt]);
      }
    }

    float bn[4];
#pragma unroll
    for (int nt = 0; nt < 4; ++nt) bn[nt] = bias[nt * 16 + lrow];
#pragma unroll
    for (int nt = 0; nt < 4; ++nt) {
      float ls = 0.f, lq = 0.f;
#pragma unroll
      for (int rt = 0; rt < 4; ++rt) {
#pragma unroll
        for (int r = 0; r < 8; ++r) {
          long long row = job + rt * 16 + r + off8;
          float v = fmaxf(acc[rt][nt][r] + bn[nt], 0.f);
          if (row < E) {
            size_t idx = (size_t)row * 64 + nt * 16 + lrow;
            if (OUT_F32) ((float*)actOutV)[idx] = v;
            else         ((_Float16*)actOutV)[idx] = (_Float16)v;
            ls += v; lq += v * v;
          }
        }
      }
      part[wv][nt][lane][0] = ls;
      part[wv][nt][lane][1] = lq;
    }
  }
  __syncthreads();
  if (tid < 128) {
    const int s = tid >> 6, n = tid & 63, nt = n >> 4, lr = n & 15;
    float a = 0.f;
#pragma unroll
    for (int w = 0; w < 8; ++w) a += part[w][nt][lr][s] + part[w][nt][lr + 16][s];
    partials[(size_t)blockIdx.x * 128 + tid] = a;
  }
}

// Reduce BN stats and fold scale/shift into the NEXT layer's weights:
// W'[k][n] = scale[k]*W[k][n] (emitted in B-fragment f16 order), b'[n] = b[n] + sum_k shift[k]*W[k][n]
__global__ __launch_bounds__(64) void megnet_fold_bn(
    const float* __restrict__ partials, int nblocks, float En,
    const float* __restrict__ g, const float* __restrict__ be,
    const float* __restrict__ W, const float* __restrict__ b,
    _Float16* __restrict__ wf, float* __restrict__ bf) {
  __shared__ float sc[64], sh[64];
  const int n = threadIdx.x;
  float s = 0.f, q = 0.f;
  for (int blk = 0; blk < nblocks; ++blk) {
    s += partials[(size_t)blk * 128 + n];
    q += partials[(size_t)blk * 128 + 64 + n];
  }
  float mean = s / En;
  float var = fmaxf(q / En - mean * mean, 0.f);
  float scl = g[n] / sqrtf(var + BN_EPS);
  sc[n] = scl;
  sh[n] = be[n] - mean * scl;
  __syncthreads();
  float accb = b[n];
  for (int k = 0; k < 64; ++k) accb += sh[k] * W[k * 64 + n];
  bf[n] = accb;
  for (int i = n; i < 4096; i += 64) {
    int h = i & 15, l = (i >> 4) & 31, f = i >> 9;
    int ks = f >> 2, nt = f & 3;
    int k = ks * 32 + ((l < 16) ? 0 : 8) + h + (h & 8);
    int nn = nt * 16 + (l & 15);
    wf[i] = (_Float16)(sc[k] * W[k * 64 + nn]);
  }
}

// Final BN: produce per-column scale/shift only.
__global__ __launch_bounds__(64) void megnet_finalize_bn(
    const float* __restrict__ partials, int nblocks, float En,
    const float* __restrict__ g, const float* __restrict__ be,
    float* __restrict__ sc_out, float* __restrict__ sh_out) {
  const int n = threadIdx.x;
  float s = 0.f, q = 0.f;
  for (int blk = 0; blk < nblocks; ++blk) {
    s += partials[(size_t)blk * 128 + n];
    q += partials[(size_t)blk * 128 + 64 + n];
  }
  float mean = s / En;
  float var = fmaxf(q / En - mean * mean, 0.f);
  float scl = g[n] / sqrtf(var + BN_EPS);
  sc_out[n] = scl;
  sh_out[n] = be[n] - mean * scl;
}

// out = act3 * scale3[col] + shift3[col], vectorized float4.
__global__ __launch_bounds__(256) void megnet_affine(
    const float* __restrict__ act3, const float* __restrict__ sc,
    const float* __restrict__ sh, float* __restrict__ out, long long total) {
  long long i = ((long long)blockIdx.x * 256 + threadIdx.x) * 4;
  if (i >= total) return;
  float4 v = *(const float4*)(act3 + i);
  int c = (int)(i & 63);
  float4 s = *(const float4*)(sc + c);
  float4 t = *(const float4*)(sh + c);
  float4 r;
  r.x = v.x * s.x + t.x;
  r.y = v.y * s.y + t.y;
  r.z = v.z * s.z + t.z;
  r.w = v.w * s.w + t.w;
  *(float4*)(out + i) = r;
}

extern "C" void kernel_launch(void* const* d_in, const int* in_sizes, int n_in,
                              void* d_out, int out_size, void* d_ws, size_t ws_size,
                              hipStream_t stream) {
  (void)n_in; (void)out_size; (void)ws_size;
  const float* src  = (const float*)d_in[0];
  const float* dst  = (const float*)d_in[1];
  const float* edg  = (const float*)d_in[2];
  const float* gat  = (const float*)d_in[3];
  const int*   bat  = (const int*)d_in[4];
  const float* W1 = (const float*)d_in[5];  const float* b1 = (const float*)d_in[6];
  const float* W2 = (const float*)d_in[7];  const float* b2 = (const float*)d_in[8];
  const float* W3 = (const float*)d_in[9];  const float* b3 = (const float*)d_in[10];
  const float* g1 = (const float*)d_in[11]; const float* be1 = (const float*)d_in[12];
  const float* g2 = (const float*)d_in[13]; const float* be2 = (const float*)d_in[14];
  const float* g3 = (const float*)d_in[15]; const float* be3 = (const float*)d_in[16];

  const int E = in_sizes[0] / 64;
  const int jobs = (E + 63) / 64;
  const int nblocks = (jobs + 7) / 8;

  char* ws = (char*)d_ws;
  const size_t EA = (size_t)E * 64;
  _Float16* act1 = (_Float16*)ws;               // f16, overlays act3 region (dead before act3 written)
  float*    act3 = (float*)ws;                  // f32, EA floats
  _Float16* act2 = (_Float16*)(ws + EA * 4);    // f16, EA halves
  char* sm = ws + EA * 4 + EA * 2;
  float* partials = (float*)sm;                 // nblocks * 128 floats
  size_t psz = (((size_t)nblocks * 128 * 4) + 255) & ~(size_t)255;
  _Float16* wf1 = (_Float16*)(sm + psz);        // 16384 halves
  _Float16* wf2 = wf1 + 16384;                  // 4096 halves
  _Float16* wf3 = wf2 + 4096;                   // 4096 halves
  float* bf2 = (float*)(wf3 + 4096);
  float* bf3 = bf2 + 64;
  float* sc3 = bf3 + 64;
  float* sh3 = sc3 + 64;

  megnet_prep_w1<<<1, 256, 0, stream>>>(W1, wf1);
  megnet_gemm1<<<nblocks, 256, 0, stream>>>(src, dst, edg, gat, bat, wf1, b1, act1, partials, E);
  megnet_fold_bn<<<1, 64, 0, stream>>>(partials, nblocks, (float)E, g1, be1, W2, b2, wf2, bf2);
  megnet_gemm_mid<false><<<nblocks, 256, 0, stream>>>(act1, wf2, bf2, (void*)act2, partials, E);
  megnet_fold_bn<<<1, 64, 0, stream>>>(partials, nblocks, (float)E, g2, be2, W3, b3, wf3, bf3);
  megnet_gemm_mid<true><<<nblocks, 256, 0, stream>>>(act2, wf3, bf3, (void*)act3, partials, E);
  megnet_finalize_bn<<<1, 64, 0, stream>>>(partials, nblocks, (float)E, g3, be3, sc3, sh3);
  const long long total = (long long)E * 64;
  const int ablocks = (int)((total / 4 + 255) / 256);
  megnet_affine<<<ablocks, 256, 0, stream>>>(act3, sc3, sh3, (float*)d_out, total);
}
